// AttentionLayer_13039520711112
// MI455X (gfx1250) — matrix operand order repaired and checked by reading, not compiled, not run on hardware
//
#include <hip/hip_runtime.h>
#include <hip/hip_bf16.h>

typedef __attribute__((ext_vector_type(16))) _Float16 v16h;
typedef __attribute__((ext_vector_type(8)))  float    v8f;

#define B_   4
#define S_   4096
#define DM_  1024
#define H_   16
#define HD_  64
#define MTOT (B_ * S_)

__device__ __forceinline__ float phi_act(float t) {
    // elu(t) + 1 == (t > 0) ? t + 1 : exp(t)
    return t > 0.0f ? t + 1.0f : __expf(t);
}

// ---- async global -> LDS staging (CDNA5, ASYNCcnt) ----------------------
// GV form: per-lane 64-bit global address (one-shot staging).
__device__ __forceinline__ void async_copy16(void* lds_dst, const void* gsrc) {
    unsigned lds = (unsigned)(size_t)lds_dst;   // low 32b of generic LDS ptr = LDS addr
    asm volatile("global_load_async_to_lds_b128 %0, %1, off"
                 :: "v"(lds), "v"((unsigned long long)(size_t)gsrc)
                 : "memory");
}
// GVS form: uniform SGPR base + 32-bit per-lane byte offset (loop staging).
__device__ __forceinline__ void async_ldg(unsigned lds, unsigned voff, const void* sbase) {
    asm volatile("global_load_async_to_lds_b128 %0, %1, %2"
                 :: "v"(lds), "v"(voff), "s"((unsigned long long)(size_t)sbase)
                 : "memory");
}
__device__ __forceinline__ void async_wait0() {
    asm volatile("s_wait_asynccnt 0x0" ::: "memory");
}

// ---- WMMA fragment loaders: all contiguous 16-byte LDS reads ------------
union F16x16 { uint4 u[2]; v16h h; };

// A 16x32 f16 fragment from row-major LDS tile [m][k]; lda in halfs (mult of 8).
__device__ __forceinline__ v16h frag_a_rowmajor(const _Float16* p0, int lda, int lane) {
    const int m  = lane & 15;
    const int hi = (lane >> 4) << 3;          // 0 or 8
    const _Float16* p = p0 + m * lda;
    F16x16 f;
    f.u[0] = *(const uint4*)(p + hi);
    f.u[1] = *(const uint4*)(p + 16 + hi);
    return f.h;
}

// B 32x16 fragment from LDS stored K-contiguous per column: T[n*ldn + k].
__device__ __forceinline__ v16h frag_b_kfast(const _Float16* p0, int ldn, int lane) {
    const _Float16* p = p0 + (lane & 15) * ldn + ((lane >> 4) << 3);
    F16x16 f;
    f.u[0] = *(const uint4*)(p);
    f.u[1] = *(const uint4*)(p + 16);
    return f.h;
}

// ---------------------------------------------------------------------------
// One-time converts: x -> f16, W -> f16 transposed [n][k]
// ---------------------------------------------------------------------------
__global__ __launch_bounds__(256) void cvt_x(const float* __restrict__ x,
                                             _Float16* __restrict__ xh) {
    const size_t i = ((size_t)blockIdx.x * 256 + threadIdx.x) * 8;
    float4 a = *(const float4*)(x + i);
    float4 b = *(const float4*)(x + i + 4);
    union { uint4 u; _Float16 h[8]; } o;
    o.h[0] = (_Float16)a.x; o.h[1] = (_Float16)a.y;
    o.h[2] = (_Float16)a.z; o.h[3] = (_Float16)a.w;
    o.h[4] = (_Float16)b.x; o.h[5] = (_Float16)b.y;
    o.h[6] = (_Float16)b.z; o.h[7] = (_Float16)b.w;
    *(uint4*)(xh + i) = o.u;
}

__global__ __launch_bounds__(256) void cvt_w_t(const float* __restrict__ W,
                                               _Float16* __restrict__ WT) {
    __shared__ _Float16 Ts[32][33];
    const int k0 = blockIdx.y * 32, n0 = blockIdx.x * 32;
    const int t = threadIdx.x;
    {
        const int k = t >> 3, c = (t & 7) * 4;
        float4 v = *(const float4*)&W[(size_t)(k0 + k) * DM_ + n0 + c];
        Ts[k][c + 0] = (_Float16)v.x; Ts[k][c + 1] = (_Float16)v.y;
        Ts[k][c + 2] = (_Float16)v.z; Ts[k][c + 3] = (_Float16)v.w;
    }
    __syncthreads();
    {
        const int n = t >> 3, c = (t & 7) * 4;
        union { unsigned long long u; _Float16 h[4]; } o;
        o.h[0] = Ts[c + 0][n]; o.h[1] = Ts[c + 1][n];
        o.h[2] = Ts[c + 2][n]; o.h[3] = Ts[c + 3][n];
        *(unsigned long long*)&WT[(size_t)(n0 + n) * DM_ + k0 + c] = o.u;
    }
}

// ---------------------------------------------------------------------------
// 128x128-tile GEMM: Out = act(A[M,K] @ WT[N,K]^T + bias)
// Explicit 2-stage software pipeline; 8 waves, each 64x32 = 4x2 WMMA tiles.
// ---------------------------------------------------------------------------
template<bool PHI, bool OUT_F32, bool OUT_TRANS>
__global__ __launch_bounds__(256) void gemm128(const _Float16* __restrict__ A,
                                               const _Float16* __restrict__ WT,
                                               const float* __restrict__ bias,
                                               void* __restrict__ Outp,
                                               int M, int N, int K) {
    __shared__ __align__(16) _Float16 As[2][128 * 32];   // [m][k]
    __shared__ __align__(16) _Float16 Bs[2][128 * 32];   // [n][k]

    const int t    = threadIdx.x;
    const int lane = t & 31;
    const int w    = t >> 5;
    const int wr   = w >> 2;                  // 0..1
    const int wc   = w & 3;                   // 0..3
    const int m0   = blockIdx.y * 128;
    const int n0   = blockIdx.x * 128;

    v8f acc[4][2];
#pragma unroll
    for (int i = 0; i < 4; ++i)
#pragma unroll
        for (int j = 0; j < 2; ++j) acc[i][j] = (v8f){0,0,0,0,0,0,0,0};

    const int srow = t >> 2;                  // 0..63
    const int sc   = (t & 3) * 8;             // 0..24

    unsigned ldsA[2], ldsB[2];
    ldsA[0] = (unsigned)(size_t)&As[0][srow * 32 + sc];
    ldsA[1] = (unsigned)(size_t)&As[1][srow * 32 + sc];
    ldsB[0] = (unsigned)(size_t)&Bs[0][srow * 32 + sc];
    ldsB[1] = (unsigned)(size_t)&Bs[1][srow * 32 + sc];
    unsigned voffA = (unsigned)(((size_t)(m0 + srow) * K + sc) * sizeof(_Float16));
    unsigned voffB = (unsigned)(((size_t)(n0 + srow) * K + sc) * sizeof(_Float16));
    const unsigned rstride = (unsigned)(64u * (unsigned)K * sizeof(_Float16)); // +64 rows

    auto issue = [&](int buf) {
        async_ldg(ldsA[buf],        voffA,           A);
        async_ldg(ldsA[buf] + 4096, voffA + rstride, A);   // 64 rows * 32 halfs
        async_ldg(ldsB[buf],        voffB,           WT);
        async_ldg(ldsB[buf] + 4096, voffB + rstride, WT);
        voffA += 64;  voffB += 64;                         // next 32-half K chunk
    };

    auto compute = [&](const _Float16* Asb, const _Float16* Bsb) {
        v16h af[4], bf[2];
#pragma unroll
        for (int i = 0; i < 4; ++i)
            af[i] = frag_a_rowmajor(Asb + (wr * 64 + i * 16) * 32, 32, lane);
#pragma unroll
        for (int j = 0; j < 2; ++j)
            bf[j] = frag_b_kfast(Bsb + (wc * 32 + j * 16) * 32, 32, lane);
#pragma unroll
        for (int i = 0; i < 4; ++i)
#pragma unroll
            for (int j = 0; j < 2; ++j)
                acc[i][j] = __builtin_amdgcn_wmma_f32_16x16x32_f16(
                    false, af[i], false, bf[j], (short)0, acc[i][j], false, false);
    };

    // 2-deep pipeline, unrolled by 2 (KCH even), conditional-free steady state
    const int KCH = K >> 5;
    issue(0);
    for (int kc = 0; kc < KCH - 2; kc += 2) {
        async_wait0(); __syncthreads(); issue(1); compute(As[0], Bs[0]);
        async_wait0(); __syncthreads(); issue(0); compute(As[1], Bs[1]);
    }
    async_wait0(); __syncthreads(); issue(1); compute(As[0], Bs[0]);
    async_wait0(); __syncthreads();           compute(As[1], Bs[1]);

    const int nl  = lane & 15;
    const int rhi = (lane >> 4) * 8;
#pragma unroll
    for (int i = 0; i < 4; ++i) {
#pragma unroll
        for (int j = 0; j < 2; ++j) {
            const int col  = n0 + wc * 32 + j * 16 + nl;
            const int row0 = m0 + wr * 64 + i * 16 + rhi;
            const float bb = bias[col];
            if constexpr (OUT_TRANS) {
                union { uint4 u; _Float16 h[8]; } o;
#pragma unroll
                for (int r = 0; r < 8; ++r) {
                    float v = acc[i][j][r] + bb;
                    if constexpr (PHI) v = phi_act(v);
                    o.h[r] = (_Float16)v;
                }
                *(uint4*)&((_Float16*)Outp)[(size_t)col * M + row0] = o.u;
            } else {
#pragma unroll
                for (int r = 0; r < 8; ++r) {
                    float v = acc[i][j][r] + bb;
                    if constexpr (PHI) v = phi_act(v);
                    if constexpr (OUT_F32)
                        ((float*)Outp)[(size_t)(row0 + r) * N + col] = v;
                    else
                        ((_Float16*)Outp)[(size_t)(row0 + r) * N + col] = (_Float16)v;
                }
            }
        }
    }
}

// ---------------------------------------------------------------------------
// kv state: per (b,h), kv[d,e] = sum_s phiK[s,d] * V[s,e]
// phiKT/VT are [DM][MTOT] f16 (pre-transposed) -> contiguous tiles.
// Explicit 2-stage pipeline. Output transposed kvT[bh][e][d].
// ---------------------------------------------------------------------------
__global__ __launch_bounds__(256) void kv_state(const _Float16* __restrict__ phiKT,
                                                const _Float16* __restrict__ VT,
                                                _Float16* __restrict__ kvT) {
    __shared__ __align__(16) _Float16 Ka[2][64 * 32];   // [d][s]
    __shared__ __align__(16) _Float16 Va[2][64 * 32];   // [e][s]

    const int t    = threadIdx.x;
    const int lane = t & 31;
    const int w    = t >> 5;
    const int bh   = blockIdx.x;
    const int b    = bh >> 4;
    const int h    = bh & 15;
    const size_t base = (size_t)h * HD_ * MTOT + (size_t)b * S_;
    const _Float16* kbase = phiKT + base;   // uniform per block
    const _Float16* vbase = VT + base;

    const int tr  = w >> 1;          // 0..3
    const int tc0 = (w & 1) * 2;     // col tiles tc0, tc0+1

    v8f acc[2];
    acc[0] = (v8f){0,0,0,0,0,0,0,0};
    acc[1] = (v8f){0,0,0,0,0,0,0,0};

    const int drow = t >> 2;         // 0..63
    const int sc   = (t & 3) * 8;    // 0..24

    unsigned ldsK[2], ldsV[2];
    ldsK[0] = (unsigned)(size_t)&Ka[0][drow * 32 + sc];
    ldsK[1] = (unsigned)(size_t)&Ka[1][drow * 32 + sc];
    ldsV[0] = (unsigned)(size_t)&Va[0][drow * 32 + sc];
    ldsV[1] = (unsigned)(size_t)&Va[1][drow * 32 + sc];
    unsigned voff = (unsigned)(((size_t)drow * MTOT + sc) * sizeof(_Float16));

    auto issue = [&](int buf) {
        async_ldg(ldsK[buf], voff, kbase);
        async_ldg(ldsV[buf], voff, vbase);
        voff += 64;                   // next 32-half chunk along s
    };

    auto compute = [&](const _Float16* Kb, const _Float16* Vb) {
        v16h a = frag_a_rowmajor(Kb + (tr * 16) * 32, 32, lane);
#pragma unroll
        for (int j = 0; j < 2; ++j) {
            v16h bf = frag_b_kfast(Vb + ((tc0 + j) * 16) * 32, 32, lane);
            acc[j] = __builtin_amdgcn_wmma_f32_16x16x32_f16(
                false, a, false, bf, (short)0, acc[j], false, false);
        }
    };

    const int NCH = S_ >> 5;          // 128 chunks (even)
    issue(0);
    for (int kc = 0; kc < NCH - 2; kc += 2) {
        async_wait0(); __syncthreads(); issue(1); compute(Ka[0], Va[0]);
        async_wait0(); __syncthreads(); issue(0); compute(Ka[1], Va[1]);
    }
    async_wait0(); __syncthreads(); issue(1); compute(Ka[0], Va[0]);
    async_wait0(); __syncthreads();           compute(Ka[1], Va[1]);

    const int nl  = lane & 15;
    const int rhi = (lane >> 4) * 8;
#pragma unroll
    for (int j = 0; j < 2; ++j) {
        const int col  = (tc0 + j) * 16 + nl;  // e
        const int row0 = tr * 16 + rhi;        // d (8 consecutive)
        union { uint4 u; _Float16 h[8]; } o;
#pragma unroll
        for (int r = 0; r < 8; ++r) o.h[r] = (_Float16)acc[j][r];
        *(uint4*)&kvT[(size_t)bh * HD_ * HD_ + (size_t)col * HD_ + row0] = o.u;
    }
}

// ---------------------------------------------------------------------------
// attn = phiQ @ kv (per b,h; M=S, K=N=64); kvT already [e][d] -> direct copy.
// ---------------------------------------------------------------------------
__global__ __launch_bounds__(256) void attn_apply(const _Float16* __restrict__ phiQ,
                                                  const _Float16* __restrict__ kvT,
                                                  _Float16* __restrict__ attn) {
    __shared__ __align__(16) _Float16 Qs[128 * 64];  // [s][d]
    __shared__ __align__(16) _Float16 KVs[64 * 64];  // [e][d]

    const int t    = threadIdx.x;
    const int lane = t & 31;
    const int w    = t >> 5;
    const int bh   = blockIdx.y;
    const int b    = bh >> 4;
    const int h    = bh & 15;
    const int s0   = blockIdx.x * 128;
    const size_t qbase = (size_t)b * S_ * DM_ + (size_t)h * HD_;

#pragma unroll
    for (int p = 0; p < 4; ++p) {
        const int row = p * 32 + (t >> 3);
        const int c   = (t & 7) * 8;
        async_copy16(&Qs[row * 64 + c], &phiQ[qbase + (size_t)(s0 + row) * DM_ + c]);
    }
    async_copy16(&KVs[t * 16],     &kvT[(size_t)bh * HD_ * HD_ + t * 16]);
    async_copy16(&KVs[t * 16 + 8], &kvT[(size_t)bh * HD_ * HD_ + t * 16 + 8]);
    async_wait0();
    __syncthreads();

    v8f acc[4];
#pragma unroll
    for (int j = 0; j < 4; ++j) acc[j] = (v8f){0,0,0,0,0,0,0,0};

#pragma unroll
    for (int kk = 0; kk < 64; kk += 32) {
        v16h a = frag_a_rowmajor(&Qs[(w * 16) * 64 + kk], 64, lane);
#pragma unroll
        for (int j = 0; j < 4; ++j) {
            v16h bf = frag_b_kfast(&KVs[(j * 16) * 64 + kk], 64, lane);
            acc[j] = __builtin_amdgcn_wmma_f32_16x16x32_f16(
                false, a, false, bf, (short)0, acc[j], false, false);
        }
    }

    const int nl  = lane & 15;
    const int rhi = (lane >> 4) * 8;
#pragma unroll
    for (int j = 0; j < 4; ++j)
#pragma unroll
        for (int r = 0; r < 8; ++r) {
            const int srow = s0 + w * 16 + rhi + r;
            const int col  = h * HD_ + j * 16 + nl;
            attn[(size_t)(b * S_ + srow) * DM_ + col] = (_Float16)acc[j][r];
        }
}

// ---------------------------------------------------------------------------
extern "C" void kernel_launch(void* const* d_in, const int* in_sizes, int n_in,
                              void* d_out, int out_size, void* d_ws, size_t ws_size,
                              hipStream_t stream) {
    const float* x  = (const float*)d_in[0];
    const float* Wq = (const float*)d_in[1];
    const float* bq = (const float*)d_in[2];
    const float* Wk = (const float*)d_in[3];
    const float* bk = (const float*)d_in[4];
    const float* Wv = (const float*)d_in[5];
    const float* bv = (const float*)d_in[6];
    const float* Wo = (const float*)d_in[7];
    const float* bo = (const float*)d_in[8];
    float* out = (float*)d_out;

    char* ws = (char*)d_ws;
    const size_t big = (size_t)MTOT * DM_ * sizeof(_Float16);    // 32 MB
    const size_t wsz = (size_t)DM_ * DM_ * sizeof(_Float16);     // 2 MB
    _Float16* xh    = (_Float16*)(ws);
    _Float16* phiQ  = (_Float16*)(ws + 1 * big);
    _Float16* phiKT = (_Float16*)(ws + 2 * big);  // [DM][MTOT]; reused as attn
    _Float16* VT    = (_Float16*)(ws + 3 * big);  // [DM][MTOT]
    _Float16* WqT   = (_Float16*)(ws + 4 * big);
    _Float16* WkT   = (_Float16*)(ws + 4 * big + 1 * wsz);
    _Float16* WvT   = (_Float16*)(ws + 4 * big + 2 * wsz);
    _Float16* WoT   = (_Float16*)(ws + 4 * big + 3 * wsz);
    _Float16* kvT   = (_Float16*)(ws + 4 * big + 4 * wsz);       // 512 KB
    _Float16* attn  = phiKT;                      // phiKT dead after kv_state

    dim3 blk(256);
    dim3 grdW(32, 32);
    dim3 grdG(DM_ / 128, MTOT / 128);             // (8, 128)

    cvt_x  <<<(MTOT * DM_) / (256 * 8), blk, 0, stream>>>(x, xh);
    cvt_w_t<<<grdW, blk, 0, stream>>>(Wq, WqT);
    cvt_w_t<<<grdW, blk, 0, stream>>>(Wk, WkT);
    cvt_w_t<<<grdW, blk, 0, stream>>>(Wv, WvT);
    cvt_w_t<<<grdW, blk, 0, stream>>>(Wo, WoT);

    gemm128<true,  false, false><<<grdG, blk, 0, stream>>>(xh, WqT, bq, phiQ,  MTOT, DM_, DM_);
    gemm128<true,  false, true ><<<grdG, blk, 0, stream>>>(xh, WkT, bk, phiKT, MTOT, DM_, DM_);
    gemm128<false, false, true ><<<grdG, blk, 0, stream>>>(xh, WvT, bv, VT,    MTOT, DM_, DM_);

    kv_state  <<<dim3(B_ * H_),           blk, 0, stream>>>(phiKT, VT, kvT);
    attn_apply<<<dim3(S_ / 128, B_ * H_), blk, 0, stream>>>(phiQ, kvT, attn);

    gemm128<false, true,  false><<<grdG, blk, 0, stream>>>(attn, WoT, bo, out, MTOT, DM_, DM_);
}
